// SSDDetectionOutput_45071386804459
// MI455X (gfx1250) — matrix operand order repaired and checked by reading, not compiled, not run on hardware
//
#include <hip/hip_runtime.h>
#include <hip/hip_bf16.h>

typedef __attribute__((ext_vector_type(16))) __bf16 v16bf;
typedef __attribute__((ext_vector_type(8)))  __bf16 v8bf;
typedef __attribute__((ext_vector_type(8)))  float  v8f;

#define NUM_PRIORS 8732
#define NUM_CLASSES 21
#define BATCH 32
#define TOPK 200
#define GEMM_WAVES 4   // waves per block
#define GEMM_MT    2   // M-tiles per wave (independent accumulators)

// ---------------------------------------------------------------------------
// NCHW fp32 -> NHWC bf16 via LDS-tiled transpose (coalesced on both sides)
// ---------------------------------------------------------------------------
__global__ void __launch_bounds__(1024)
nchw2nhwc_bf16(const float* __restrict__ in, __bf16* __restrict__ out, int C, int HW) {
    __shared__ float tile[32][33];
    const int b  = blockIdx.z;
    const int hw0 = blockIdx.x * 32;
    const int c0  = blockIdx.y * 32;
    const int tx = threadIdx.x, ty = threadIdx.y;
    const int c  = c0 + ty, hw = hw0 + tx;
    if (c < C && hw < HW) tile[ty][tx] = in[((size_t)b * C + c) * HW + hw];
    __syncthreads();
    const int hw2 = hw0 + ty, c2 = c0 + tx;
    if (hw2 < HW && c2 < C)
        out[((size_t)b * HW + hw2) * C + c2] = (__bf16)tile[tx][ty];
}

// ---------------------------------------------------------------------------
// Repack loc+conf conv weights [O][Cin][3][3] fp32 -> Bt[Npad][K] bf16,
// k = (ky*3+kx)*Cin + ci ; n = [loc channels | conf channels | zero pad]
// ---------------------------------------------------------------------------
__global__ void repack_weights(const float* __restrict__ locw, const float* __restrict__ confw,
                               __bf16* __restrict__ Bt, int npc, int Cin, int K, int Npad) {
    size_t i = (size_t)blockIdx.x * blockDim.x + threadIdx.x;
    size_t total = (size_t)Npad * K;
    if (i >= total) return;
    int k = (int)(i % K);
    int n = (int)(i / K);
    int tap = k / Cin;
    int ci  = k - tap * Cin;
    int nloc = npc * 4, nconf = npc * NUM_CLASSES;
    float v = 0.f;
    if (n < nloc)               v = locw [((size_t)n * Cin + ci) * 9 + tap];
    else if (n < nloc + nconf)  v = confw[((size_t)(n - nloc) * Cin + ci) * 9 + tap];
    Bt[i] = (__bf16)v;
}

// ---------------------------------------------------------------------------
// Stage one 16(cols)x32(k) bf16 B-chunk (1KB) into LDS, 16B per thread,
// using the gfx1250 async global->LDS DMA path when the builtin exists.
// Builtin signature (from compiler diagnostic): b128 variant takes pointers
// to 16-byte int vectors in AS1 (global) / AS3 (LDS), plus imm offset, cpol.
// ---------------------------------------------------------------------------
#if __has_builtin(__builtin_amdgcn_global_load_async_to_lds_b128)
typedef int async_v4i __attribute__((vector_size(16)));
typedef __attribute__((address_space(1))) async_v4i* g_v4i_ptr;
typedef __attribute__((address_space(3))) async_v4i* l_v4i_ptr;
#endif

__device__ __forceinline__ void stage_b_chunk(const __bf16* __restrict__ Bt, int K,
                                              int n0, int k0, __bf16* dstBase, int t) {
    if (t < 64) {
        const int n_l = t >> 2, part = t & 3;
        const __bf16* src = Bt + (size_t)(n0 + n_l) * K + k0 + part * 8;
        __bf16* dst = dstBase + n_l * 32 + part * 8;
#if __has_builtin(__builtin_amdgcn_global_load_async_to_lds_b128)
        __builtin_amdgcn_global_load_async_to_lds_b128(
            (g_v4i_ptr)(uintptr_t)src,
            (l_v4i_ptr)(uint32_t)(uintptr_t)dst,
            0, 0);
#else
        *(v8bf*)dst = *(const v8bf*)src;
#endif
    }
}

__device__ __forceinline__ void wait_async_done() {
#if __has_builtin(__builtin_amdgcn_s_wait_asynccnt)
    __builtin_amdgcn_s_wait_asynccnt(0);
#else
#if __has_builtin(__builtin_amdgcn_global_load_async_to_lds_b128)
    asm volatile("s_wait_asynccnt 0" ::: "memory");
#endif
#endif
}

// ---------------------------------------------------------------------------
// Implicit-GEMM conv head.
// Block = 4 waves; each wave owns 2 consecutive 16x16 M-tiles (independent
// accumulators -> back-to-back v_wmma with no RAW hazard), all waves share
// one tileN column. B chunks are double-buffered in LDS via async DMA.
// Outer loop over the 9 conv taps hoists all boundary math; inner loop over
// Cin (step 32) is loads + ds_load + 2x v_wmma_f32_16x16x32_bf16 only.
// A/B fragments use contiguous k = 16*half + e; C layout per CDNA5 ISA 7.12.2 (wave32).
// ---------------------------------------------------------------------------
__global__ void __launch_bounds__(GEMM_WAVES * 32)
wmma_conv_gemm(const __bf16* __restrict__ featT, const __bf16* __restrict__ Bt,
               float* __restrict__ C, int H, int W, int Cin, int K, int Npad, int Mtot) {
    __shared__ __attribute__((aligned(64))) __bf16 lbs[2][16 * 32];

    const int t    = threadIdx.x;
    const int wave = t >> 5;
    const int lane = t & 31;
    const int half = lane >> 4;
    const int lm   = lane & 15;
    const int n0   = blockIdx.y * 16;

    const int tile0 = (blockIdx.x * GEMM_WAVES + wave) * GEMM_MT;
    const bool v0 = (tile0 * 16)      < Mtot;   // uniform per wave
    const bool v1 = (tile0 * 16 + 16) < Mtot;

    const int m0 = tile0 * 16 + lm;
    const int m1 = m0 + 16;
    const int x0 = m0 % W, q0 = m0 / W, y0 = q0 % H, b0 = q0 / H;
    const int x1 = m1 % W, q1 = m1 / W, y1 = q1 % H, b1 = q1 / H;

    v8f acc0 = {}, acc1 = {};
    int ck = 0;
    stage_b_chunk(Bt, K, n0, 0, &lbs[0][0], t);

    for (int tap = 0; tap < 9; ++tap) {
        const int ky = tap / 3, kx = tap - ky * 3;        // fully unrolled consts
        const int yy0 = y0 + ky - 1, xx0 = x0 + kx - 1;
        const int yy1 = y1 + ky - 1, xx1 = x1 + kx - 1;
        const bool in0 = v0 && yy0 >= 0 && yy0 < H && xx0 >= 0 && xx0 < W;
        const bool in1 = v1 && yy1 >= 0 && yy1 < H && xx1 >= 0 && xx1 < W;
        const __bf16* ap0 = featT + (((size_t)b0 * H + yy0) * W + xx0) * Cin + half * 16;
        const __bf16* ap1 = featT + (((size_t)b1 * H + yy1) * W + xx1) * Cin + half * 16;
        const int kbase = tap * Cin;

        for (int ci = 0; ci < Cin; ci += 32, ++ck) {
            wait_async_done();
            __syncthreads();                               // lbs[ck&1] ready
            const int knext = kbase + ci + 32;
            if (knext < K) stage_b_chunk(Bt, K, n0, knext, &lbs[(ck + 1) & 1][0], t);

            // B fragment: lane column lm, k = half*16 + e (32B contiguous LDS)
            const v16bf bfr = *(const v16bf*)(&lbs[ck & 1][0] + lm * 32 + half * 16);

            // A fragments: k = half*16 + e
            v8bf lo0 = {}, hi0 = {}, lo1 = {}, hi1 = {};
            if (in0) { lo0 = *(const v8bf*)(ap0 + ci); hi0 = *(const v8bf*)(ap0 + ci + 8); }
            if (in1) { lo1 = *(const v8bf*)(ap1 + ci); hi1 = *(const v8bf*)(ap1 + ci + 8); }
            v16bf a0, a1;
#pragma unroll
            for (int e = 0; e < 8; ++e) {
                a0[e] = lo0[e]; a0[e + 8] = hi0[e];
                a1[e] = lo1[e]; a1[e + 8] = hi1[e];
            }

            acc0 = __builtin_amdgcn_wmma_f32_16x16x32_bf16(
                       false, a0, false, bfr, (short)0, acc0, false, false);
            acc1 = __builtin_amdgcn_wmma_f32_16x16x32_bf16(
                       false, a1, false, bfr, (short)0, acc1, false, false);
            __syncthreads();                               // done reading lbs[ck&1]
        }
    }

    // C store: lane half h, acc element v -> row = 8h+v, col = lm
    if (v0) {
        float* crow = C + (size_t)(tile0 * 16 + half * 8) * Npad + n0 + lm;
#pragma unroll
        for (int v = 0; v < 8; ++v) crow[(size_t)v * Npad] = acc0[v];
    }
    if (v1) {
        float* crow = C + (size_t)(tile0 * 16 + 16 + half * 8) * Npad + n0 + lm;
#pragma unroll
        for (int v = 0; v < 8; ++v) crow[(size_t)v * Npad] = acc1[v];
    }
}

// ---------------------------------------------------------------------------
// Caffe-style PriorBox generation (matches _make_priors concatenation order)
// ---------------------------------------------------------------------------
__global__ void gen_priors(float* __restrict__ priors) {
    int p = blockIdx.x * blockDim.x + threadIdx.x;
    if (p >= NUM_PRIORS) return;
    const int   fw_[6]   = {38, 19, 10, 5, 3, 1};
    const int   npc_[6]  = {4, 6, 6, 6, 4, 4};
    const float mins[6]  = {30.f, 60.f, 111.f, 162.f, 213.f, 264.f};
    const float maxs[6]  = {60.f, 111.f, 162.f, 213.f, 264.f, 315.f};
    const float steps[6] = {8.f, 16.f, 32.f, 64.f, 100.f, 300.f};
    int s = 0, rem = p;
    for (; s < 6; ++s) { int cnt = fw_[s] * fw_[s] * npc_[s]; if (rem < cnt) break; rem -= cnt; }
    const int npc = npc_[s], fw = fw_[s];
    const int j = rem % npc, cell = rem / npc;
    const int cx_i = cell % fw, cy_i = cell / fw;
    const float m = mins[s], M = maxs[s], st = steps[s];
    const float cx = (cx_i + 0.5f) * st / 300.f;
    const float cy = (cy_i + 0.5f) * st / 300.f;
    float w, h;
    if (j == 0)      { w = m; h = m; }
    else if (j == 1) { float q = sqrtf(m * M); w = q; h = q; }
    else {
        float r = sqrtf((j < 4) ? 2.f : 3.f);
        if ((j & 1) == 0) { w = m * r; h = m / r; } else { w = m / r; h = m * r; }
    }
    w /= 300.f; h /= 300.f;
    float* o = priors + (size_t)p * 4;
    o[0] = cx - w * 0.5f; o[1] = cy - h * 0.5f;
    o[2] = cx + w * 0.5f; o[3] = cy + h * 0.5f;
}

// ---------------------------------------------------------------------------
// Per (b, position, prior): add bias, softmax(21), decode box with variance
// ---------------------------------------------------------------------------
__global__ void head_assemble(const float* __restrict__ C, const float* __restrict__ locb,
                              const float* __restrict__ confb, const float* __restrict__ priors,
                              float* __restrict__ confOut, float* __restrict__ boxOut,
                              int HWtot, int npc, int Npad, int scaleOff) {
    int i = blockIdx.x * blockDim.x + threadIdx.x;
    int total = BATCH * HWtot * npc;
    if (i >= total) return;
    const int j = i % npc;
    int t = i / npc;
    const int pos = t % HWtot;
    const int b   = t / HWtot;
    const float* crow = C + ((size_t)b * HWtot + pos) * Npad;
    const int p = scaleOff + pos * npc + j;

    float logit[NUM_CLASSES], mx = -1e30f;
#pragma unroll
    for (int c = 0; c < NUM_CLASSES; ++c) {
        float v = crow[npc * 4 + j * NUM_CLASSES + c] + confb[j * NUM_CLASSES + c];
        logit[c] = v; mx = fmaxf(mx, v);
    }
    float sum = 0.f;
#pragma unroll
    for (int c = 0; c < NUM_CLASSES; ++c) { logit[c] = __expf(logit[c] - mx); sum += logit[c]; }
    const float inv = 1.f / sum;
    float* co = confOut + ((size_t)b * NUM_PRIORS + p) * (NUM_CLASSES - 1);
#pragma unroll
    for (int c = 1; c < NUM_CLASSES; ++c) co[c - 1] = logit[c] * inv;

    const float lx = crow[j * 4 + 0] + locb[j * 4 + 0];
    const float ly = crow[j * 4 + 1] + locb[j * 4 + 1];
    const float lw = crow[j * 4 + 2] + locb[j * 4 + 2];
    const float lh = crow[j * 4 + 3] + locb[j * 4 + 3];
    const float* pr = priors + (size_t)p * 4;
    const float pw = pr[2] - pr[0], ph = pr[3] - pr[1];
    const float pcx = (pr[0] + pr[2]) * 0.5f, pcy = (pr[1] + pr[3]) * 0.5f;
    const float cx = pcx + 0.1f * lx * pw;
    const float cy = pcy + 0.1f * ly * ph;
    const float w  = pw * __expf(0.2f * lw);
    const float h  = ph * __expf(0.2f * lh);
    float* bo = boxOut + ((size_t)b * NUM_PRIORS + p) * 4;
    bo[0] = cx - w * 0.5f; bo[1] = cy - h * 0.5f;
    bo[2] = cx + w * 0.5f; bo[3] = cy + h * 0.5f;
}

// ---------------------------------------------------------------------------
// One block per (image, class): threshold, iterative top-200, greedy NMS
// ---------------------------------------------------------------------------
__global__ void __launch_bounds__(256)
nms_per_class(const float* __restrict__ conf, const float* __restrict__ boxes,
              float* __restrict__ clsScore, float* __restrict__ clsBox) {
    const int b = blockIdx.x / (NUM_CLASSES - 1);
    const int c = blockIdx.x % (NUM_CLASSES - 1);
    const int tid = threadIdx.x;

    __shared__ float s[NUM_PRIORS];
    __shared__ float redV[256];
    __shared__ int   redI[256];
    __shared__ float topV[TOPK];
    __shared__ int   topI[TOPK];
    __shared__ float bx[TOPK * 4];
    __shared__ int   keep[TOPK];

    for (int p = tid; p < NUM_PRIORS; p += 256) {
        float v = conf[((size_t)b * NUM_PRIORS + p) * (NUM_CLASSES - 1) + c];
        s[p] = (v > 0.01f) ? v : 0.f;
    }
    __syncthreads();

    for (int r = 0; r < TOPK; ++r) {
        float bv = 0.f; int bi = -1;
        for (int p = tid; p < NUM_PRIORS; p += 256) {
            float v = s[p];
            if (v > bv) { bv = v; bi = p; }
        }
        redV[tid] = bv; redI[tid] = bi;
        __syncthreads();
        for (int off = 128; off > 0; off >>= 1) {
            if (tid < off) {
                float ov = redV[tid + off]; int oi = redI[tid + off];
                if (ov > redV[tid] || (ov == redV[tid] && oi >= 0 &&
                                       (redI[tid] < 0 || oi < redI[tid]))) {
                    redV[tid] = ov; redI[tid] = oi;
                }
            }
            __syncthreads();
        }
        if (tid == 0) {
            topV[r] = redV[0]; topI[r] = redI[0];
            if (redI[0] >= 0) s[redI[0]] = -1.f;
        }
        __syncthreads();
    }

    if (tid < TOPK) {
        int i = topI[tid];
        if (i >= 0) {
#pragma unroll
            for (int d = 0; d < 4; ++d)
                bx[tid * 4 + d] = boxes[((size_t)b * NUM_PRIORS + i) * 4 + d];
        } else {
#pragma unroll
            for (int d = 0; d < 4; ++d) bx[tid * 4 + d] = 0.f;
        }
        keep[tid] = (topV[tid] > 0.f) ? 1 : 0;
    }
    __syncthreads();

    for (int i = 0; i < TOPK - 1; ++i) {
        if (keep[i] && tid > i && tid < TOPK && keep[tid]) {
            float x1 = fmaxf(bx[i * 4 + 0], bx[tid * 4 + 0]);
            float y1 = fmaxf(bx[i * 4 + 1], bx[tid * 4 + 1]);
            float x2 = fminf(bx[i * 4 + 2], bx[tid * 4 + 2]);
            float y2 = fminf(bx[i * 4 + 3], bx[tid * 4 + 3]);
            float inter = fmaxf(x2 - x1, 0.f) * fmaxf(y2 - y1, 0.f);
            float ai = (bx[i * 4 + 2] - bx[i * 4 + 0]) * (bx[i * 4 + 3] - bx[i * 4 + 1]);
            float at = (bx[tid * 4 + 2] - bx[tid * 4 + 0]) * (bx[tid * 4 + 3] - bx[tid * 4 + 1]);
            float un = fmaxf(ai + at - inter, 1e-9f);
            if (inter / un > 0.45f) keep[tid] = 0;
        }
        __syncthreads();
    }

    if (tid < TOPK) {
        size_t o = ((size_t)b * (NUM_CLASSES - 1) + c) * TOPK + tid;
        clsScore[o] = keep[tid] ? topV[tid] : 0.f;
#pragma unroll
        for (int d = 0; d < 4; ++d) clsBox[o * 4 + d] = bx[tid * 4 + d];
    }
}

// ---------------------------------------------------------------------------
// One block per image: top-200 of 20*200 candidates -> [200,7] rows
// ---------------------------------------------------------------------------
__global__ void __launch_bounds__(256)
final_topk(const float* __restrict__ clsScore, const float* __restrict__ clsBox,
           float* __restrict__ out) {
    const int b = blockIdx.x;
    const int tid = threadIdx.x;
    const int N = (NUM_CLASSES - 1) * TOPK;   // 4000
    __shared__ float fs[(NUM_CLASSES - 1) * TOPK];
    __shared__ float redV[256];
    __shared__ int   redI[256];

    for (int i = tid; i < N; i += 256) fs[i] = clsScore[(size_t)b * N + i];
    __syncthreads();

    for (int r = 0; r < TOPK; ++r) {
        float bv = 0.f; int bi = -1;
        for (int i = tid; i < N; i += 256) {
            float v = fs[i];
            if (v > bv) { bv = v; bi = i; }
        }
        redV[tid] = bv; redI[tid] = bi;
        __syncthreads();
        for (int off = 128; off > 0; off >>= 1) {
            if (tid < off) {
                float ov = redV[tid + off]; int oi = redI[tid + off];
                if (ov > redV[tid] || (ov == redV[tid] && oi >= 0 &&
                                       (redI[tid] < 0 || oi < redI[tid]))) {
                    redV[tid] = ov; redI[tid] = oi;
                }
            }
            __syncthreads();
        }
        if (tid == 0) {
            float v = redV[0]; int i = redI[0];
            float* o = out + ((size_t)b * TOPK + r) * 7;
            o[0] = (float)b;
            if (i >= 0) {
                fs[i] = -1.f;
                o[1] = (v > 0.f) ? (float)(i / TOPK + 1) : -1.f;
                o[2] = (v > 0.f) ? v : 0.f;
#pragma unroll
                for (int d = 0; d < 4; ++d)
                    o[3 + d] = clsBox[((size_t)b * N + i) * 4 + d];
            } else {
                o[1] = -1.f; o[2] = 0.f;
                o[3] = 0.f; o[4] = 0.f; o[5] = 0.f; o[6] = 0.f;
            }
        }
        __syncthreads();
    }
}

// ---------------------------------------------------------------------------
extern "C" void kernel_launch(void* const* d_in, const int* in_sizes, int n_in,
                              void* d_out, int out_size, void* d_ws, size_t ws_size,
                              hipStream_t stream) {
    (void)in_sizes; (void)n_in; (void)out_size;
    static const int CH[6]  = {512, 1024, 512, 256, 256, 256};
    static const int HWs[6] = {38, 19, 10, 5, 3, 1};
    static const int NPC[6] = {4, 6, 6, 6, 4, 4};
    static const int OFF[6] = {0, 5776, 7942, 8542, 8692, 8728};

    char* ws = (char*)d_ws;
    size_t cur = 0;
    auto alloc = [&](size_t bytes) -> char* {
        char* p = ws + cur;
        cur = (cur + bytes + 255) & ~(size_t)255;
        return p;
    };

    float* priors   = (float*)alloc((size_t)NUM_PRIORS * 4 * sizeof(float));
    float* confBuf  = (float*)alloc((size_t)BATCH * NUM_PRIORS * (NUM_CLASSES - 1) * sizeof(float));
    float* boxBuf   = (float*)alloc((size_t)BATCH * NUM_PRIORS * 4 * sizeof(float));
    float* clsScore = (float*)alloc((size_t)BATCH * (NUM_CLASSES - 1) * TOPK * sizeof(float));
    float* clsBox   = (float*)alloc((size_t)BATCH * (NUM_CLASSES - 1) * TOPK * 4 * sizeof(float));

    __bf16* featT[6]; __bf16* Bt[6]; float* Cbuf[6];
    int Npad[6];
    for (int i = 0; i < 6; ++i) {
        const int HW = HWs[i] * HWs[i];
        const int K  = CH[i] * 9;
        Npad[i] = ((NPC[i] * 25 + 15) / 16) * 16;
        featT[i] = (__bf16*)alloc((size_t)BATCH * HW * CH[i] * sizeof(__bf16));
        Bt[i]    = (__bf16*)alloc((size_t)Npad[i] * K * sizeof(__bf16));
        Cbuf[i]  = (float*)alloc((size_t)BATCH * HW * Npad[i] * sizeof(float));
    }
    if (cur > ws_size) return;  // workspace too small: bail rather than corrupt

    gen_priors<<<(NUM_PRIORS + 255) / 256, 256, 0, stream>>>(priors);

    for (int i = 0; i < 6; ++i) {
        const float* feat  = (const float*)d_in[i];
        const float* locw  = (const float*)d_in[7 + i * 4 + 0];
        const float* locb  = (const float*)d_in[7 + i * 4 + 1];
        const float* confw = (const float*)d_in[7 + i * 4 + 2];
        const float* confb = (const float*)d_in[7 + i * 4 + 3];
        const int H = HWs[i], W = HWs[i], Cin = CH[i], K = Cin * 9;
        const int HW = H * W, M = BATCH * HW;

        dim3 tgrid((HW + 31) / 32, (Cin + 31) / 32, BATCH);
        nchw2nhwc_bf16<<<tgrid, dim3(32, 32), 0, stream>>>(feat, featT[i], Cin, HW);

        size_t tot2 = (size_t)Npad[i] * K;
        repack_weights<<<(unsigned)((tot2 + 255) / 256), 256, 0, stream>>>(
            locw, confw, Bt[i], NPC[i], Cin, K, Npad[i]);

        const int tilesM  = M / 16;                        // all M multiples of 16
        const int tilesPB = GEMM_WAVES * GEMM_MT;          // 8 M-tiles per block
        dim3 grid((tilesM + tilesPB - 1) / tilesPB, Npad[i] / 16);
        wmma_conv_gemm<<<grid, GEMM_WAVES * 32, 0, stream>>>(
            featT[i], Bt[i], Cbuf[i], H, W, Cin, K, Npad[i], M);

        int tot3 = BATCH * HW * NPC[i];
        head_assemble<<<(tot3 + 255) / 256, 256, 0, stream>>>(
            Cbuf[i], locb, confb, priors, confBuf, boxBuf, HW, NPC[i], Npad[i], OFF[i]);
    }

    nms_per_class<<<BATCH * (NUM_CLASSES - 1), 256, 0, stream>>>(confBuf, boxBuf, clsScore, clsBox);
    final_topk<<<BATCH, 256, 0, stream>>>(clsScore, clsBox, (float*)d_out);
}
